// MultiHeadAttention_90692529423029
// MI455X (gfx1250) — matrix operand order repaired and checked by reading, not compiled, not run on hardware
//
#include <hip/hip_runtime.h>
#include <hip/hip_bf16.h>

// ---------------------------------------------------------------------------
// MHA forward for B=2, L=2048, D_MODEL=1024, H=16, D_HEAD=64 (fp32 in/out).
// Matmuls on v_wmma_f32_16x16x32_f16 (f32 accumulate). Intermediates (qkv,
// attn) stored as f16 to halve HBM/L2 traffic; K-tile and out-proj A-tile
// staging uses global_load_async_to_lds_b128 (ASYNCcnt) on gfx1250.
// ---------------------------------------------------------------------------

#define BATCH 2
#define SEQ   2048
#define DMODEL 1024
#define NHEAD 16
#define DHEAD 64

typedef __attribute__((ext_vector_type(16))) _Float16 v16h;
typedef __attribute__((ext_vector_type(8)))  _Float16 v8h;
typedef __attribute__((ext_vector_type(8)))  float    v8f;

__device__ __forceinline__ v16h frag16(const _Float16* p0, const _Float16* p1) {
    v8h a = *(const v8h*)p0;   // ds_load_b128
    v8h b = *(const v8h*)p1;   // ds_load_b128
    return __builtin_shufflevector(a, b, 0,1,2,3,4,5,6,7,8,9,10,11,12,13,14,15);
}

__device__ __forceinline__ v8f wmma16(v16h a, v16h b, v8f c) {
    return __builtin_amdgcn_wmma_f32_16x16x32_f16(
        /*neg_a=*/false, a, /*neg_b=*/false, b,
        /*c_mod=*/(short)0, c, /*reuse_a=*/false, /*reuse_b=*/false);
}

// Async global->LDS 16-byte copy (gfx1250 GLOBAL_LOAD_ASYNC_TO_LDS_B128).
__device__ __forceinline__ void async_copy_b128(const _Float16* gptr, _Float16* lptr) {
    unsigned           lds = (unsigned)(unsigned long long)lptr;   // LDS aperture: addr[31:0]
    unsigned long long ga  = (unsigned long long)gptr;
    asm volatile("global_load_async_to_lds_b128 %0, %1, off" :: "v"(lds), "v"(ga) : "memory");
}
__device__ __forceinline__ void wait_async0() {
    asm volatile("s_wait_asynccnt 0x0" ::: "memory");
}

// ---------------------------------------------------------------------------
// C = A[M,K] @ W(f32)[K,N] + bias. TA in {float, _Float16}; TC likewise.
// Block = 256 threads = 8 waves. Tile: 128(M) x 64(N), K-step 32.
// ---------------------------------------------------------------------------
template <typename TA, typename TC>
__global__ __launch_bounds__(256) void gemm_bias_wmma(
    const TA* __restrict__ A, const float* __restrict__ W,
    const float* __restrict__ bias, TC* __restrict__ C,
    int M, int N, int K)
{
    constexpr bool A16 = (sizeof(TA) == 2);
    constexpr bool C16 = (sizeof(TC) == 2);

    __shared__ __align__(16) _Float16 As[128 * 32]; // [m][k], row stride 32
    __shared__ __align__(16) _Float16 Bs[64 * 32];  // [n][k] = W tile transposed

    const int tid  = threadIdx.x;
    const int wave = tid >> 5;
    const int lane = tid & 31;
    const int n16  = lane & 15;
    const int hi   = lane >> 4;
    const int m0   = blockIdx.y * 128;
    const int n0   = blockIdx.x * 64;

    v8f acc[4] = {};

    for (int k0 = 0; k0 < K; k0 += 32) {
        __syncthreads();
        if constexpr (A16) {
            // Pure copy: 512 x 16B chunks, 2 per thread, via async-to-LDS
            #pragma unroll
            for (int j = 0; j < 2; ++j) {
                int c = tid + 256 * j;              // chunk id
                int r = c >> 2, kk = (c & 3) * 8;   // 4 chunks per 32-half row
                async_copy_b128((const _Float16*)&A[(m0 + r) * K + k0 + kk],
                                &As[r * 32 + kk]);
            }
        } else {
            // f32 -> f16 convert path (16 per thread), coalesced on k
            #pragma unroll
            for (int j = 0; j < 16; ++j) {
                int idx = tid + 256 * j;            // idx = r*32 + k
                int r = idx >> 5, k = idx & 31;
                As[idx] = (_Float16)(float)A[(m0 + r) * K + k0 + k];
            }
        }
        // B tile transposed: 32x64 of W -> Bs[n][k] (8 per thread), coalesced on n
        #pragma unroll
        for (int j = 0; j < 8; ++j) {
            int idx = tid + 256 * j;
            int k = idx >> 6, n = idx & 63;
            Bs[n * 32 + k] = (_Float16)W[(k0 + k) * N + n0 + n];
        }
        if constexpr (A16) wait_async0();
        __syncthreads();

        // A fragment (16x32): lane m=n16; elem i -> K = hi*16 + i
        const int mr = wave * 16 + n16;
        v16h af = frag16(&As[mr * 32 + hi * 16], &As[mr * 32 + hi * 16 + 8]);
        #pragma unroll
        for (int t = 0; t < 4; ++t) {
            // B fragment (32x16): lane column n; elem i -> K = hi*16 + i
            const int nr = t * 16 + n16;
            v16h bf = frag16(&Bs[nr * 32 + hi * 16], &Bs[nr * 32 + hi * 16 + 8]);
            acc[t] = wmma16(af, bf, acc[t]);
        }
    }

    // Epilogue: D tile -> lane (n16,hi), VGPR r => row r+8*hi, col n16
    #pragma unroll
    for (int t = 0; t < 4; ++t) {
        const int gn = n0 + t * 16 + n16;
        const float bv = bias[gn];
        #pragma unroll
        for (int r = 0; r < 8; ++r) {
            const int gm = m0 + wave * 16 + r + 8 * hi;
            if constexpr (C16) C[gm * N + gn] = (_Float16)(acc[t][r] + bv);
            else               C[gm * N + gn] = acc[t][r] + bv;
        }
    }
}

// ---------------------------------------------------------------------------
// Flash attention over f16 qkv: [B*L, 3*DMODEL], col = sel*1024 + h*64 + d.
// Block = 128 threads (4 waves); wave owns 16 queries; block = 64 queries of
// one (b,h). Key blocks of 32. Writes f16 attn [B*L, DMODEL].
// ---------------------------------------------------------------------------
__global__ __launch_bounds__(128) void flash_attn_wmma(
    const _Float16* __restrict__ qkv, _Float16* __restrict__ out)
{
    __shared__ __align__(16) _Float16 Kt[32 * 64];     // [key][d]
    __shared__ __align__(16) _Float16 Vt[64 * 32];     // [d][key] (transposed)
    __shared__ __align__(16) _Float16 Qt[4][16 * 64];  // per-wave [row][d]
    __shared__ __align__(16) _Float16 Pt[4][16 * 32];  // per-wave [row][key]

    const int tid  = threadIdx.x;
    const int wave = tid >> 5;
    const int lane = tid & 31;
    const int n16  = lane & 15;
    const int hi   = lane >> 4;
    const int b    = blockIdx.z;
    const int h    = blockIdx.y;
    const int q0   = blockIdx.x * 64 + wave * 16;

    // 1/sqrt(DHEAD) * log2(e) folded into Q so softmax uses exp2
    const float QS = 0.125f * 1.44269504088896f;

    // Stage this wave's 16x64 Q tile (scaled): 128 chunks of 8 halves, 4/lane
    #pragma unroll
    for (int j = 0; j < 4; ++j) {
        int c = lane + 32 * j;
        int r = c >> 3, dp = (c & 7) * 8;
        v8h q = *(const v8h*)&qkv[(b * SEQ + q0 + r) * (3 * DMODEL) + h * DHEAD + dp];
        #pragma unroll
        for (int e = 0; e < 8; ++e) q[e] = (_Float16)((float)q[e] * QS);
        *(v8h*)&Qt[wave][r * 64 + dp] = q;
    }
    __syncthreads();

    // Q A-fragments for K-dim steps d=0..31 and d=32..63 (live across loop)
    const int m = n16;
    v16h qa0 = frag16(&Qt[wave][m * 64 +      hi * 16], &Qt[wave][m * 64 +  8 + hi * 16]);
    v16h qa1 = frag16(&Qt[wave][m * 64 + 32 + hi * 16], &Qt[wave][m * 64 + 40 + hi * 16]);

    v8f acc0 = {}, acc1 = {}, acc2 = {}, acc3 = {};
    float mrow[8], lrow[8];
    #pragma unroll
    for (int r = 0; r < 8; ++r) { mrow[r] = -1e30f; lrow[r] = 0.0f; }

    for (int kb = 0; kb < SEQ; kb += 32) {
        __syncthreads();
        // K tile: 256 x 16B pure copies -> async to LDS (2 per thread)
        #pragma unroll
        for (int j = 0; j < 2; ++j) {
            int c = tid + 128 * j;
            int key = c >> 3, dp = (c & 7) * 8;
            async_copy_b128(&qkv[(b * SEQ + kb + key) * (3 * DMODEL) + DMODEL + h * DHEAD + dp],
                            &Kt[key * 64 + dp]);
        }
        // V tile transposed: load 8-half chunks, scatter b16 into Vt[d][key]
        #pragma unroll
        for (int j = 0; j < 2; ++j) {
            int c = tid + 128 * j;
            int key = c >> 3, dp = (c & 7) * 8;
            v8h v = *(const v8h*)&qkv[(b * SEQ + kb + key) * (3 * DMODEL) + 2 * DMODEL + h * DHEAD + dp];
            #pragma unroll
            for (int e = 0; e < 8; ++e) Vt[(dp + e) * 32 + key] = v[e];
        }
        wait_async0();
        __syncthreads();

        // S = Q @ K^T : two 16x16 score tiles (keys kb..+15, kb+16..+31)
        v8f s0 = {}, s1 = {};
        {
            v16h k00 = frag16(&Kt[n16 * 64 +      hi * 16], &Kt[n16 * 64 +      hi * 16 + 8]);
            v16h k01 = frag16(&Kt[n16 * 64 + 32 + hi * 16], &Kt[n16 * 64 + 32 + hi * 16 + 8]);
            s0 = wmma16(qa0, k00, s0);
            s0 = wmma16(qa1, k01, s0);
            v16h k10 = frag16(&Kt[(16 + n16) * 64 +      hi * 16], &Kt[(16 + n16) * 64 +      hi * 16 + 8]);
            v16h k11 = frag16(&Kt[(16 + n16) * 64 + 32 + hi * 16], &Kt[(16 + n16) * 64 + 32 + hi * 16 + 8]);
            s1 = wmma16(qa0, k10, s1);
            s1 = wmma16(qa1, k11, s1);
        }

        // Online softmax (row = r + 8*hi; reduce across 16 lanes of same half)
        #pragma unroll
        for (int r = 0; r < 8; ++r) {
            float part = fmaxf(s0[r], s1[r]);
            part = fmaxf(part, __shfl_xor(part, 1));
            part = fmaxf(part, __shfl_xor(part, 2));
            part = fmaxf(part, __shfl_xor(part, 4));
            part = fmaxf(part, __shfl_xor(part, 8));
            float nm = fmaxf(mrow[r], part);
            float cf = exp2f(mrow[r] - nm);
            mrow[r] = nm;
            float p0 = exp2f(s0[r] - nm);
            float p1 = exp2f(s1[r] - nm);
            float rs = p0 + p1;
            rs += __shfl_xor(rs, 1);
            rs += __shfl_xor(rs, 2);
            rs += __shfl_xor(rs, 4);
            rs += __shfl_xor(rs, 8);
            lrow[r] = lrow[r] * cf + rs;
            acc0[r] *= cf; acc1[r] *= cf; acc2[r] *= cf; acc3[r] *= cf;
            // Re-layout P: D-tile (row r+8*hi, col n16 / n16+16) -> LDS [row][key]
            Pt[wave][(r + 8 * hi) * 32 + n16]      = (_Float16)p0;
            Pt[wave][(r + 8 * hi) * 32 + 16 + n16] = (_Float16)p1;
        }
        // Wave-private LDS store->load; keep compiler & HW ordered
        asm volatile("s_wait_dscnt 0x0" ::: "memory");

        // P A-fragment (16x32 over keys) and O += P @ V
        v16h pa = frag16(&Pt[wave][n16 * 32 + hi * 16], &Pt[wave][n16 * 32 + hi * 16 + 8]);
        {
            v16h vb0 = frag16(&Vt[( 0 + n16) * 32 + hi * 16], &Vt[( 0 + n16) * 32 + hi * 16 + 8]);
            acc0 = wmma16(pa, vb0, acc0);
            v16h vb1 = frag16(&Vt[(16 + n16) * 32 + hi * 16], &Vt[(16 + n16) * 32 + hi * 16 + 8]);
            acc1 = wmma16(pa, vb1, acc1);
            v16h vb2 = frag16(&Vt[(32 + n16) * 32 + hi * 16], &Vt[(32 + n16) * 32 + hi * 16 + 8]);
            acc2 = wmma16(pa, vb2, acc2);
            v16h vb3 = frag16(&Vt[(48 + n16) * 32 + hi * 16], &Vt[(48 + n16) * 32 + hi * 16 + 8]);
            acc3 = wmma16(pa, vb3, acc3);
        }
    }

    // Normalize and write f16 [B*L, DMODEL] (col = h*64 + d)
    #pragma unroll
    for (int r = 0; r < 8; ++r) {
        float inv = 1.0f / lrow[r];
        int row = b * SEQ + q0 + r + 8 * hi;
        int col = h * DHEAD + n16;
        out[row * DMODEL + col     ] = (_Float16)(acc0[r] * inv);
        out[row * DMODEL + col + 16] = (_Float16)(acc1[r] * inv);
        out[row * DMODEL + col + 32] = (_Float16)(acc2[r] * inv);
        out[row * DMODEL + col + 48] = (_Float16)(acc3[r] * inv);
    }
}

extern "C" void kernel_launch(void* const* d_in, const int* in_sizes, int n_in,
                              void* d_out, int out_size, void* d_ws, size_t ws_size,
                              hipStream_t stream) {
    const float* x     = (const float*)d_in[0];   // [2,2048,1024]
    const float* W_qkv = (const float*)d_in[1];   // [1024,3072]
    const float* b_qkv = (const float*)d_in[2];   // [3072]
    const float* W_o   = (const float*)d_in[3];   // [1024,1024]
    const float* b_o   = (const float*)d_in[4];   // [1024]
    float* outp = (float*)d_out;

    const int M = BATCH * SEQ;                    // 4096
    _Float16* qkv16  = (_Float16*)d_ws;                        // 4096*3072 f16 = 24 MB
    _Float16* attn16 = qkv16 + (size_t)M * 3 * DMODEL;         // 4096*1024 f16 =  8 MB

    // 1) QKV projection: f32 in -> f16 out (bias added in f32)
    gemm_bias_wmma<float, _Float16>
        <<<dim3((3 * DMODEL) / 64, M / 128), 256, 0, stream>>>(
            x, W_qkv, b_qkv, qkv16, M, 3 * DMODEL, DMODEL);

    // 2) Flash attention per (b, h), 64 queries per block
    flash_attn_wmma<<<dim3(SEQ / 64, NHEAD, BATCH), 128, 0, stream>>>(
        qkv16, attn16);

    // 3) Output projection: f16 in -> f32 out into d_out
    gemm_bias_wmma<_Float16, float>
        <<<dim3(DMODEL / 64, M / 128), 256, 0, stream>>>(
            attn16, W_o, b_o, outp, M, DMODEL, DMODEL);
}
